// LocalGNN_9689446219791
// MI455X (gfx1250) — compile-verified, matrix operand order checked by reading, not run on hardware
//
#include <hip/hip_runtime.h>
#include <hip/hip_bf16.h>
#include <math.h>

// ---------------------------------------------------------------------------
// EGNN layer stack for MI455X (gfx1250): bf16 WMMA GEMMs, f32 accumulate.
// Round 2: 2 M-tiles per wave (B-fragment reuse x2), packed b128 LDS staging,
// dynamic LDS (>64KB static limit; CDNA5 allows 320KB/WGP).
// ---------------------------------------------------------------------------

typedef __attribute__((ext_vector_type(16))) __bf16 v16bf;
typedef __attribute__((ext_vector_type(8)))  float  v8f;

#define HDIM   128
#define ASTR_E 328      // LDS row stride (bf16 elems) for edge A staging (292->320 pad)
#define K1N    264      // LDS row stride for node A staging (256 pad)
#define HSTR   136      // LDS row stride for hidden (K=128) staging
#define XSTR   132      // f32 LN staging stride (aliases K1N bf16 rows)

#define EDGE_SMEM (128 * ASTR_E * 2 + 128 * HSTR * 2 + 128 * 4)   // 119,296 B
#define NODE_SMEM (128 * K1N * 2 + 128 * HSTR * 2)                 // 102,400 B

// pack two floats into 2x bf16 in one dword
__device__ __forceinline__ unsigned pk2(float a, float b) {
    union { __bf16 h[2]; unsigned u; } x;
    x.h[0] = (__bf16)a; x.h[1] = (__bf16)b;
    return x.u;
}

// pack 8 floats (two float4) into a uint4 of 8 bf16
__device__ __forceinline__ uint4 pk8(float4 a, float4 b) {
    uint4 u;
    u.x = pk2(a.x, a.y); u.y = pk2(a.z, a.w);
    u.z = pk2(b.x, b.y); u.w = pk2(b.z, b.w);
    return u;
}

// 16B-aligned load of a 32B bf16x16 fragment (two ds/global b128 loads)
__device__ __forceinline__ v16bf ld_frag(const __bf16* p) {
    union { uint4 q[2]; v16bf v; } u;
    u.q[0] = *(const uint4*)(p);
    u.q[1] = *(const uint4*)(p + 8);
    return u.v;
}

__device__ __forceinline__ v8f wmma_bf16(v16bf a, v16bf b, v8f c) {
    return __builtin_amdgcn_wmma_f32_16x16x32_bf16(
        /*neg_a=*/false, a, /*neg_b=*/false, b,
        /*c_mod=*/(short)0, c, /*reuse_a=*/false, /*reuse_b=*/false);
}

// ---------------------------------------------------------------------------
// Weight prepack: W[k][n] fp32 (row-major, N=128) -> bf16 in WMMA-B lane
// layout: dst[(((nt*ksteps)+ks)*32 + lane)*16 + j],
//   n = nt*16 + (lane&15), k = ks*32 + (lane>>4)*16 + j. K padded with zeros.
// ---------------------------------------------------------------------------
__global__ void gnn_prepack_w(const float* __restrict__ W, __bf16* __restrict__ dst,
                              int Korig, int ksteps, int total) {
    int i = blockIdx.x * 256 + threadIdx.x;
    if (i >= total) return;
    int j    = i & 15;
    int lane = (i >> 4) & 31;
    int rest = i >> 9;
    int ks = rest % ksteps;
    int nt = rest / ksteps;
    int n = nt * 16 + (lane & 15);
    int k = ks * 32 + (lane >> 4) * 16 + j;
    float v = (k < Korig) ? W[(size_t)k * HDIM + n] : 0.0f;
    dst[i] = (__bf16)v;
}

__global__ void gnn_zero_f32(float* __restrict__ p, long long n) {
    long long i = (long long)blockIdx.x * 1024 + threadIdx.x;
    if (i < n) p[i] = 0.0f;
}

// ---------------------------------------------------------------------------
// Edge kernel: 128 edges/block, 4 waves; each wave owns 32 edges (2 M-tiles)
// so every B fragment feeds two WMMAs.
// msg = silu([h[row],h[col],ea,dist,unit] @ W1 + b1) @ W2 + b2 ; atomic scatter.
// ---------------------------------------------------------------------------
__global__ __launch_bounds__(128)
void gnn_edge_kernel(const float* __restrict__ h, const float* __restrict__ pos,
                     const int* __restrict__ ei, const float* __restrict__ ea,
                     const __bf16* __restrict__ w1, const float* __restrict__ b1,
                     const __bf16* __restrict__ w2, const float* __restrict__ b2,
                     float* __restrict__ agg, int Etot) {
    extern __shared__ char smem[];
    __bf16* Abuf   = (__bf16*)smem;                                    // 128 x 328
    __bf16* Hbuf   = (__bf16*)(smem + 128 * ASTR_E * 2);               // 128 x 136
    int*    colbuf = (int*)   (smem + 128 * ASTR_E * 2 + 128 * HSTR * 2);

    const int tid  = threadIdx.x;
    const int base = blockIdx.x * 128;

    // ---- stage msg_in rows: one thread per edge, packed b128 stores ----
    {
        int r = tid;
        int e = base + r; if (e >= Etot) e = Etot - 1;
        int erow = ei[e];
        int ecol = ei[Etot + e];
        uint4* dst = (uint4*)(&Abuf[r * ASTR_E]);   // 40 uint4 = 320 elems

        const float* hr = &h[(size_t)erow * HDIM];
        const float* hc = &h[(size_t)ecol * HDIM];
        #pragma unroll
        for (int i = 0; i < 16; ++i)
            dst[i]      = pk8(((const float4*)hr)[2 * i], ((const float4*)hr)[2 * i + 1]);
        #pragma unroll
        for (int i = 0; i < 16; ++i)
            dst[16 + i] = pk8(((const float4*)hc)[2 * i], ((const float4*)hc)[2 * i + 1]);

        const float* ap = &ea[(size_t)e * 32];
        #pragma unroll
        for (int i = 0; i < 4; ++i)
            dst[32 + i] = pk8(((const float4*)ap)[2 * i], ((const float4*)ap)[2 * i + 1]);

        float dx = pos[3 * erow + 0] - pos[3 * ecol + 0];
        float dy = pos[3 * erow + 1] - pos[3 * ecol + 1];
        float dz = pos[3 * erow + 2] - pos[3 * ecol + 2];
        float d  = fmaxf(sqrtf(dx * dx + dy * dy + dz * dz), 1e-6f);
        float inv = 1.0f / d;
        uint4 g; g.x = pk2(d, dx * inv); g.y = pk2(dy * inv, dz * inv); g.z = 0u; g.w = 0u;
        dst[36] = g;
        uint4 z; z.x = 0u; z.y = 0u; z.z = 0u; z.w = 0u;
        dst[37] = z; dst[38] = z; dst[39] = z;     // pad 296..319

        colbuf[r] = ecol;
    }
    __syncthreads();

    const int lane = tid & 31, wave = tid >> 5;
    const int lh = lane >> 4, ln = lane & 15;

    // ---- GEMM1: 2 x [16 x 320] @ [320 x 128], B reused for both tiles ----
    v8f acc0[8] = {}, acc1[8] = {};
    const __bf16* A0 = &Abuf[(wave * 32 + ln) * ASTR_E];
    const __bf16* A1 = A0 + 16 * ASTR_E;
    #pragma unroll
    for (int k = 0; k < 10; ++k) {
        v16bf a0 = ld_frag(A0 + k * 32 + lh * 16);
        v16bf a1 = ld_frag(A1 + k * 32 + lh * 16);
        #pragma unroll
        for (int nt = 0; nt < 8; ++nt) {
            v16bf b = *(const v16bf*)(w1 + ((size_t)((nt * 10 + k) * 32 + lane)) * 16);
            acc0[nt] = wmma_bf16(a0, b, acc0[nt]);
            acc1[nt] = wmma_bf16(a1, b, acc1[nt]);
        }
    }

    // ---- bias + SiLU, restage hidden (bf16) for both tiles ----
    __bf16* hb = &Hbuf[wave * 32 * HSTR];
    #pragma unroll
    for (int nt = 0; nt < 8; ++nt) {
        float bias = b1[nt * 16 + ln];
        #pragma unroll
        for (int v = 0; v < 8; ++v) {
            float x0 = acc0[nt][v] + bias;
            float x1 = acc1[nt][v] + bias;
            hb[(v + 8 * lh) * HSTR + nt * 16 + ln]        = (__bf16)(x0 / (1.0f + expf(-x0)));
            hb[(16 + v + 8 * lh) * HSTR + nt * 16 + ln]   = (__bf16)(x1 / (1.0f + expf(-x1)));
        }
    }
    __syncthreads();

    // ---- GEMM2: 2 x [16 x 128] @ [128 x 128] ----
    v8f acc2a[8] = {}, acc2b[8] = {};
    const __bf16* H0 = hb + ln * HSTR;
    const __bf16* H1 = hb + (16 + ln) * HSTR;
    #pragma unroll
    for (int k = 0; k < 4; ++k) {
        v16bf a0 = ld_frag(H0 + k * 32 + lh * 16);
        v16bf a1 = ld_frag(H1 + k * 32 + lh * 16);
        #pragma unroll
        for (int nt = 0; nt < 8; ++nt) {
            v16bf b = *(const v16bf*)(w2 + ((size_t)((nt * 4 + k) * 32 + lane)) * 16);
            acc2a[nt] = wmma_bf16(a0, b, acc2a[nt]);
            acc2b[nt] = wmma_bf16(a1, b, acc2b[nt]);
        }
    }

    // ---- bias + atomic scatter-add into agg[col] ----
    int cols0[8], cols1[8];
    #pragma unroll
    for (int v = 0; v < 8; ++v) {
        cols0[v] = colbuf[wave * 32 + v + 8 * lh];
        cols1[v] = colbuf[wave * 32 + 16 + v + 8 * lh];
    }
    #pragma unroll
    for (int nt = 0; nt < 8; ++nt) {
        float bias = b2[nt * 16 + ln];
        int n = nt * 16 + ln;
        #pragma unroll
        for (int v = 0; v < 8; ++v) {
            int e0 = base + wave * 32 + v + 8 * lh;
            int e1 = e0 + 16;
            if (e0 < Etot)
                unsafeAtomicAdd(&agg[(size_t)cols0[v] * HDIM + n], acc2a[nt][v] + bias);
            if (e1 < Etot)
                unsafeAtomicAdd(&agg[(size_t)cols1[v] * HDIM + n], acc2b[nt][v] + bias);
        }
    }
}

// ---------------------------------------------------------------------------
// Node kernel: 128 nodes/block, 4 waves x 2 M-tiles.
// h_new = silu([h,agg]@U1+b1)@U2+b2 ; out = LayerNorm(h + h_new).
// ---------------------------------------------------------------------------
__global__ __launch_bounds__(128)
void gnn_node_kernel(const float* __restrict__ h, const float* __restrict__ agg,
                     const __bf16* __restrict__ w1, const float* __restrict__ b1,
                     const __bf16* __restrict__ w2, const float* __restrict__ b2,
                     const float* __restrict__ lnw, const float* __restrict__ lnb,
                     float* __restrict__ out, int Ntot) {
    extern __shared__ char smem[];
    __bf16* Abuf = (__bf16*)smem;                       // 128 x 264 bf16 ...
    float*  xbuf = (float*)smem;                        // ... aliases 128 x 132 f32
    __bf16* Hbuf = (__bf16*)(smem + 128 * K1N * 2);     // 128 x 136

    const int tid  = threadIdx.x;
    const int base = blockIdx.x * 128;

    // ---- stage [h | agg] rows as bf16, one thread per node ----
    {
        int r = tid;
        int nd = base + r; if (nd >= Ntot) nd = Ntot - 1;
        uint4* dst = (uint4*)(&Abuf[r * K1N]);          // 33 uint4 = 264 elems
        const float* hp = &h[(size_t)nd * HDIM];
        const float* gp = &agg[(size_t)nd * HDIM];
        #pragma unroll
        for (int i = 0; i < 16; ++i)
            dst[i]      = pk8(((const float4*)hp)[2 * i], ((const float4*)hp)[2 * i + 1]);
        #pragma unroll
        for (int i = 0; i < 16; ++i)
            dst[16 + i] = pk8(((const float4*)gp)[2 * i], ((const float4*)gp)[2 * i + 1]);
        uint4 z; z.x = 0u; z.y = 0u; z.z = 0u; z.w = 0u;
        dst[32] = z;                                    // pad 256..263
    }
    __syncthreads();

    const int lane = tid & 31, wave = tid >> 5;
    const int lh = lane >> 4, ln = lane & 15;

    // ---- GEMM1: 2 x [16 x 256] @ [256 x 128] ----
    v8f acc0[8] = {}, acc1[8] = {};
    const __bf16* A0 = &Abuf[(wave * 32 + ln) * K1N];
    const __bf16* A1 = A0 + 16 * K1N;
    #pragma unroll
    for (int k = 0; k < 8; ++k) {
        v16bf a0 = ld_frag(A0 + k * 32 + lh * 16);
        v16bf a1 = ld_frag(A1 + k * 32 + lh * 16);
        #pragma unroll
        for (int nt = 0; nt < 8; ++nt) {
            v16bf b = *(const v16bf*)(w1 + ((size_t)((nt * 8 + k) * 32 + lane)) * 16);
            acc0[nt] = wmma_bf16(a0, b, acc0[nt]);
            acc1[nt] = wmma_bf16(a1, b, acc1[nt]);
        }
    }

    // ---- bias + SiLU, restage hidden ----
    __bf16* hb = &Hbuf[wave * 32 * HSTR];
    #pragma unroll
    for (int nt = 0; nt < 8; ++nt) {
        float bias = b1[nt * 16 + ln];
        #pragma unroll
        for (int v = 0; v < 8; ++v) {
            float x0 = acc0[nt][v] + bias;
            float x1 = acc1[nt][v] + bias;
            hb[(v + 8 * lh) * HSTR + nt * 16 + ln]      = (__bf16)(x0 / (1.0f + expf(-x0)));
            hb[(16 + v + 8 * lh) * HSTR + nt * 16 + ln] = (__bf16)(x1 / (1.0f + expf(-x1)));
        }
    }
    __syncthreads();

    // ---- GEMM2: 2 x [16 x 128] @ [128 x 128] -> h_new into xbuf (aliases Abuf) ----
    v8f acc2a[8] = {}, acc2b[8] = {};
    const __bf16* H0 = hb + ln * HSTR;
    const __bf16* H1 = hb + (16 + ln) * HSTR;
    #pragma unroll
    for (int k = 0; k < 4; ++k) {
        v16bf a0 = ld_frag(H0 + k * 32 + lh * 16);
        v16bf a1 = ld_frag(H1 + k * 32 + lh * 16);
        #pragma unroll
        for (int nt = 0; nt < 8; ++nt) {
            v16bf b = *(const v16bf*)(w2 + ((size_t)((nt * 4 + k) * 32 + lane)) * 16);
            acc2a[nt] = wmma_bf16(a0, b, acc2a[nt]);
            acc2b[nt] = wmma_bf16(a1, b, acc2b[nt]);
        }
    }
    #pragma unroll
    for (int nt = 0; nt < 8; ++nt) {
        float bias = b2[nt * 16 + ln];
        #pragma unroll
        for (int v = 0; v < 8; ++v) {
            int r0 = wave * 32 + v + 8 * lh;
            xbuf[r0 * XSTR + nt * 16 + ln]        = acc2a[nt][v] + bias;
            xbuf[(r0 + 16) * XSTR + nt * 16 + ln] = acc2b[nt][v] + bias;
        }
    }
    __syncthreads();

    // ---- residual + LayerNorm, one thread per node row ----
    {
        int nd = base + tid;
        if (nd < Ntot) {
            const float* hp = &h[(size_t)nd * HDIM];
            float* xr = &xbuf[tid * XSTR];
            float s = 0.0f, s2 = 0.0f;
            #pragma unroll 8
            for (int i = 0; i < HDIM; ++i) {
                float x = hp[i] + xr[i];
                xr[i] = x;
                s += x; s2 += x * x;
            }
            float mu  = s * (1.0f / HDIM);
            float var = s2 * (1.0f / HDIM) - mu * mu;
            float rs  = rsqrtf(var + 1e-5f);
            float* op = &out[(size_t)nd * HDIM];
            #pragma unroll 8
            for (int i = 0; i < HDIM; ++i)
                op[i] = (xr[i] - mu) * rs * lnw[i] + lnb[i];
        }
    }
}

// ---------------------------------------------------------------------------
// Host launcher
// ---------------------------------------------------------------------------
extern "C" void kernel_launch(void* const* d_in, const int* in_sizes, int n_in,
                              void* d_out, int out_size, void* d_ws, size_t ws_size,
                              hipStream_t stream) {
    const float* h_in = (const float*)d_in[0];
    const float* pos  = (const float*)d_in[1];
    const int*   ei   = (const int*)d_in[2];
    const float* ea   = (const float*)d_in[3];
    float* out = (float*)d_out;

    const int N = in_sizes[0] / HDIM;      // 100000
    const int E = in_sizes[3] / 32;        // 400000

    // workspace layout
    float* agg = (float*)d_ws;
    float* hb0 = agg + (size_t)N * HDIM;
    float* hb1 = hb0 + (size_t)N * HDIM;
    __bf16* pkbase = (__bf16*)(hb1 + (size_t)N * HDIM);

    const int M1P = 10 * 8 * 32 * 16;   // 40960
    const int M2P = 4  * 8 * 32 * 16;   // 16384
    const int U1P = 8  * 8 * 32 * 16;   // 32768
    const int U2P = 4  * 8 * 32 * 16;   // 16384
    const int LP  = M1P + M2P + U1P + U2P;

    // prepack all weights (params at d_in[4 + 10*l + k]):
    // k: 0=m1w 1=m1b 2=m2w 3=m2b 4=u1w 5=u1b 6=u2w 7=u2b 8=lnw 9=lnb
    for (int l = 0; l < 3; ++l) {
        __bf16* m1p = pkbase + (size_t)l * LP;
        __bf16* m2p = m1p + M1P;
        __bf16* u1p = m2p + M2P;
        __bf16* u2p = u1p + U1P;
        gnn_prepack_w<<<(M1P + 255) / 256, 256, 0, stream>>>(
            (const float*)d_in[4 + 10 * l + 0], m1p, 292, 10, M1P);
        gnn_prepack_w<<<(M2P + 255) / 256, 256, 0, stream>>>(
            (const float*)d_in[4 + 10 * l + 2], m2p, 128, 4, M2P);
        gnn_prepack_w<<<(U1P + 255) / 256, 256, 0, stream>>>(
            (const float*)d_in[4 + 10 * l + 4], u1p, 256, 8, U1P);
        gnn_prepack_w<<<(U2P + 255) / 256, 256, 0, stream>>>(
            (const float*)d_in[4 + 10 * l + 6], u2p, 128, 4, U2P);
    }

    const float* hc = h_in;
    float* houts[3] = { hb0, hb1, out };
    const long long aggN = (long long)N * HDIM;

    for (int l = 0; l < 3; ++l) {
        __bf16* m1p = pkbase + (size_t)l * LP;
        __bf16* m2p = m1p + M1P;
        __bf16* u1p = m2p + M2P;
        __bf16* u2p = u1p + U1P;
        const float* m1b = (const float*)d_in[4 + 10 * l + 1];
        const float* m2b = (const float*)d_in[4 + 10 * l + 3];
        const float* u1b = (const float*)d_in[4 + 10 * l + 5];
        const float* u2b = (const float*)d_in[4 + 10 * l + 7];
        const float* lnw = (const float*)d_in[4 + 10 * l + 8];
        const float* lnb = (const float*)d_in[4 + 10 * l + 9];

        gnn_zero_f32<<<(int)((aggN + 1023) / 1024), 1024, 0, stream>>>(agg, aggN);

        gnn_edge_kernel<<<(E + 127) / 128, 128, EDGE_SMEM, stream>>>(
            hc, pos, ei, ea, m1p, m1b, m2p, m2b, agg, E);

        gnn_node_kernel<<<(N + 127) / 128, 128, NODE_SMEM, stream>>>(
            hc, agg, u1p, u1b, u2p, u2b, lnw, lnb, houts[l], N);

        hc = houts[l];
    }
}
